// RNNSentiment_73383811219931
// MI455X (gfx1250) — matrix operand order repaired and checked by reading, not compiled, not run on hardware
//
#include <hip/hip_runtime.h>
#include <cmath>

// Problem constants (from reference)
#define V_   100000
#define D_   300
#define H_   512
#define B_   25
#define T_   600
#define BT_  15000          // B*T
#define MP_  15008          // BT padded to multiple of 16
#define K0P_ 320            // D padded to multiple of 32
#define G_   1536           // 3*H
#define NT_  96             // G_/16 column tiles

typedef __bf16 bf16;
typedef __attribute__((ext_vector_type(16))) __bf16 v16bf;
typedef __attribute__((ext_vector_type(8)))  __bf16 v8bf;
typedef __attribute__((ext_vector_type(8)))  float  v8f;

__device__ __forceinline__ v8f wmma_bf16(v16bf a, v16bf b, v8f c) {
    return __builtin_amdgcn_wmma_f32_16x16x32_bf16(
        false, a, false, b, (short)0, c, false, false);
}

// 16 bf16 A-fragment elements from a per-lane base pointer (row*ld + 8*half
// already applied): elements 0..7 at p, 8..15 at p+16.
__device__ __forceinline__ v16bf load_a16(const bf16* p) {
    v8bf lo = *(const v8bf*)p;
    v8bf hi = *(const v8bf*)(p + 16);
    v16bf r;
#pragma unroll
    for (int i = 0; i < 8; ++i) { r[i] = lo[i]; r[8 + i] = hi[i]; }
    return r;
}

__device__ __forceinline__ void grid_barrier(int* cnt, int& expected) {
    __syncthreads();
    expected += (int)gridDim.x;
    if (threadIdx.x == 0) {
        __hip_atomic_fetch_add(cnt, 1, __ATOMIC_RELEASE, __HIP_MEMORY_SCOPE_AGENT);
        while (__hip_atomic_load(cnt, __ATOMIC_ACQUIRE, __HIP_MEMORY_SCOPE_AGENT) < expected) {
            __builtin_amdgcn_s_sleep(1);
        }
    }
    __syncthreads();
}

__device__ __forceinline__ float sigmoidf_(float x) { return 1.f / (1.f + expf(-x)); }

// ---- Kernels ----------------------------------------------------------------

// Init: ping-pong hidden buffers (f32 + bf16 mirrors, rows>=25 zero), X1 pad, counters.
// hwork layout per layer: [2][32*H_] f32 ; HB per layer: [2][32*H_] bf16.
__global__ __launch_bounds__(256) void init_state_kernel(
    const float* __restrict__ hidden, float* __restrict__ h0w, float* __restrict__ h1w,
    bf16* __restrict__ hb0, bf16* __restrict__ hb1,
    bf16* __restrict__ X1, int* __restrict__ cnt) {
    int i = blockIdx.x * 256 + threadIdx.x;
    const int HW = 32 * H_;            // one buffer
    const int TOT = 2 * 2 * HW;        // 2 layers x 2 buffers
    if (i < TOT) {
        int l   = i / (2 * HW);
        int rem = i % (2 * HW);        // buf*HW + j
        int j   = rem % HW;
        int r   = j / H_;
        float v = (r < B_) ? hidden[l * B_ * H_ + r * H_ + (j % H_)] : 0.f;
        if (l == 0) { h0w[rem] = v; hb0[rem] = (bf16)v; }
        else        { h1w[rem] = v; hb1[rem] = (bf16)v; }
    } else if (i < TOT + (MP_ - BT_) * H_) {
        int j = i - TOT;
        X1[(size_t)BT_ * H_ + j] = (bf16)0.f;
    } else if (i < TOT + (MP_ - BT_) * H_ + 64) {
        cnt[i - (TOT + (MP_ - BT_) * H_)] = 0;
    }
}

// Swizzle W [G_ x Kreal] f32 into bf16 B-fragments [ktiles][96][512] (K zero-padded).
__global__ __launch_bounds__(256) void convert_weight_kernel(
    const float* __restrict__ W, bf16* __restrict__ F, int Kreal, int ktiles) {
    int i = blockIdx.x * 256 + threadIdx.x;
    int total = ktiles * NT_ * 512;
    if (i >= total) return;
    int e    = i & 15;
    int lane = (i >> 4) & 31;
    int frag = i >> 9;
    int nt   = frag % NT_;
    int kt   = frag / NT_;
    int n    = nt * 16 + (lane & 15);
    int k    = kt * 32 + (lane >> 4) * 8 + (e & 7) + (e >> 3) * 16;
    float v  = (k < Kreal) ? W[(size_t)n * Kreal + k] : 0.f;
    F[(size_t)frag * 512 + lane * 16 + e] = (bf16)v;
}

// Embedding gather -> padded bf16 activation matrix X0 [MP_ x K0P_].
__global__ __launch_bounds__(256) void embed_gather_kernel(
    const int* __restrict__ words, const float* __restrict__ emb,
    bf16* __restrict__ X0) {
    long i = (long)blockIdx.x * 256 + threadIdx.x;
    if (i >= (long)MP_ * K0P_) return;
    int m = (int)(i / K0P_);
    int k = (int)(i % K0P_);
    float v = 0.f;
    if (m < BT_ && k < D_) {
        int w = words[m];
        v = emb[(size_t)w * D_ + k];
    }
    X0[i] = (bf16)v;
}

// C = A @ WfragT + bias.  Block: 16 M-tiles x 4 N-tiles; the 4-N-tile B panel
// (ktiles x 4KB) is staged once into LDS and shared by all 8 waves. Each wave:
// 2 M-tiles x 4 N-tiles, 8 WMMAs per K-step (4 global b128 + 8 ds reads).
__global__ __launch_bounds__(256) void gemm_bias_kernel(
    const bf16* __restrict__ A, int lda, int ktiles,
    const bf16* __restrict__ WF, const float* __restrict__ bias,
    float* __restrict__ C, int mtiles) {
    extern __shared__ char smemg[];
    bf16* ldsB = (bf16*)smemg;               // [ktiles][4*512]
    const int wave = threadIdx.x >> 5;
    const int lane = threadIdx.x & 31;
    const int mpairs = mtiles >> 1;          // 469
    const int nt4 = blockIdx.x % (NT_ / 4);  // block's group of 4 n-tiles
    const int mp  = (blockIdx.x / (NT_ / 4)) * 8 + wave;

    // Stage the B panel: ktiles * 4KB, 16B per thread per iteration.
    for (int c = threadIdx.x; c < ktiles * 256; c += 256) {
        int kt  = c >> 8;
        int off = (c & 255) * 8;
        *(v8bf*)(ldsB + kt * 2048 + off) =
            *(const v8bf*)(WF + (size_t)(kt * NT_ + nt4 * 4) * 512 + off);
    }
    __syncthreads();
    if (mp >= mpairs) return;

    const int mt0 = mp * 2;
    const int half = lane >> 4;
    const bf16* a0 = A + (size_t)(mt0 * 16 + (lane & 15)) * lda + 8 * half;
    const bf16* a1 = a0 + (size_t)16 * lda;
    const bf16* bp = ldsB + lane * 16;

    v8f acc[2][4] = {};
    for (int kt = 0; kt < ktiles; ++kt) {
        v16bf af0 = load_a16(a0);
        v16bf af1 = load_a16(a1);
#pragma unroll
        for (int j = 0; j < 4; ++j) {
            v16bf bf = *(const v16bf*)(bp + j * 512);
            acc[0][j] = wmma_bf16(af0, bf, acc[0][j]);
            acc[1][j] = wmma_bf16(af1, bf, acc[1][j]);
        }
        a0 += 32; a1 += 32; bp += 2048;
    }
#pragma unroll
    for (int j = 0; j < 4; ++j) {
        int n = (nt4 * 4 + j) * 16 + (lane & 15);
        float bv = bias[n];
#pragma unroll
        for (int i2 = 0; i2 < 2; ++i2) {
#pragma unroll
            for (int v = 0; v < 8; ++v) {
                int m = (mt0 + i2) * 16 + v + 8 * half;
                C[(size_t)m * G_ + n] = acc[i2][j][v] + bv;
            }
        }
    }
}

// Persistent GRU scan: 8 WGs x 384 threads (12 waves). Ping-pong hidden state
// (read h[t&1], write h[(t+1)&1]) -> ONE grid barrier per timestep. Each WG
// owns 64 h-cols with its 512x192 bf16 w_hh slice in LDS; wave w owns N-tile w
// for both M-tiles (each LDS B-fragment feeds 2 WMMAs).
__global__ __launch_bounds__(384) void gru_scan_kernel(
    const float* __restrict__ xg,    // rows b*T_+t, ld = G_
    const bf16* __restrict__ WhhF,   // [16][96][512] fragment order
    const float* __restrict__ b_hh,  // [G_]
    float* __restrict__ hwork,       // [2][32][H_] f32 ping-pong
    bf16* __restrict__ HB,           // [2][32][H_] bf16 mirror
    bf16* __restrict__ outSeq,       // [MP_][H_] bf16 or nullptr
    int* __restrict__ cnt) {
    extern __shared__ char smem[];
    bf16*  wsl   = (bf16*)smem;                                   // 16*12*512 bf16
    float* gh    = (float*)(smem + 16 * 12 * 512 * sizeof(bf16)); // 32*192 f32
    float* biasL = gh + 32 * 192;                                 // 192 f32

    const int tid  = threadIdx.x;
    const int wave = tid >> 5;           // 0..11 == local n-tile
    const int lane = tid & 31;
    const int wg   = blockIdx.x;         // 0..7

    for (int i = tid; i < 16 * 12 * 512; i += 384) {
        int e2  = i & 511;
        int lnt = (i >> 9) % 12;
        int kt  = i / (12 * 512);
        int gate = lnt >> 2;
        int gnt  = gate * 32 + wg * 4 + (lnt & 3);
        wsl[i] = WhhF[(size_t)(kt * NT_ + gnt) * 512 + e2];
    }
    for (int i = tid; i < 192; i += 384)
        biasL[i] = b_hh[(i >> 6) * 512 + wg * 64 + (i & 63)];
    __syncthreads();

    const int half = lane >> 4;
    const size_t aoff = (size_t)(lane & 15) * H_ + 8 * half;
    const bf16* bbase = wsl + (size_t)wave * 512 + lane * 16;
    const int HW = 32 * H_;

    int expected = 0;
    for (int t = 0; t < T_; ++t) {
        const int cur = t & 1, nxt = cur ^ 1;
        // gh = h @ w_hh_sliceT
        {
            v8f acc0 = {}, acc1 = {};
            const bf16* a0 = HB + (size_t)cur * HW + aoff;
            const bf16* a1 = a0 + 16 * H_;
            const bf16* bp = bbase;
#pragma unroll 4
            for (int kt = 0; kt < 16; ++kt) {
                v16bf bf = *(const v16bf*)bp;
                acc0 = wmma_bf16(load_a16(a0), bf, acc0);
                acc1 = wmma_bf16(load_a16(a1), bf, acc1);
                a0 += 32; a1 += 32; bp += 12 * 512;
            }
            int col = wave * 16 + (lane & 15);
#pragma unroll
            for (int v = 0; v < 8; ++v) {
                gh[(v + 8 * half) * 192 + col]      = acc0[v];
                gh[(16 + v + 8 * half) * 192 + col] = acc1[v];
            }
        }
        __syncthreads();   // gh complete within this WG

        for (int i = tid; i < B_ * 64; i += 384) {
            int b = i >> 6, c = i & 63;
            int col = wg * 64 + c;
            long row = (long)b * T_ + t;
            const float* xr = xg + row * G_;
            float hr = gh[b * 192 + c]       + biasL[c];
            float hz = gh[b * 192 + 64 + c]  + biasL[64 + c];
            float hn = gh[b * 192 + 128 + c] + biasL[128 + c];
            float r  = sigmoidf_(xr[col] + hr);
            float z  = sigmoidf_(xr[512 + col] + hz);
            float n  = tanhf(xr[1024 + col] + r * hn);
            float hp = hwork[(size_t)cur * HW + b * H_ + col];
            float hv = (1.f - z) * n + z * hp;
            hwork[(size_t)nxt * HW + b * H_ + col] = hv;
            HB[(size_t)nxt * HW + b * H_ + col]    = (bf16)hv;
            if (outSeq) outSeq[row * H_ + col] = (bf16)hv;
            __builtin_prefetch(xr + G_, 0, 1);   // next step's xg row
        }
        // Next-buffer writes visible to all WGs before next step's GEMM.
        grid_barrier(cnt, expected);
    }
    // T_ even: final hidden state ends up in buffer 0 (read by finalize).
}

// sig = sigmoid(h1_final @ fc_wT + fc_b); then copy final hiddens (buffer 0).
__global__ __launch_bounds__(256) void finalize_kernel(
    const float* __restrict__ h0w, const float* __restrict__ h1w,
    const float* __restrict__ fcw, const float* __restrict__ fcb,
    float* __restrict__ out) {
    int i = blockIdx.x * 256 + threadIdx.x;
    if (i < B_) {
        float acc = fcb[0];
        for (int j = 0; j < H_; ++j) acc += h1w[i * H_ + j] * fcw[j];
        out[i] = sigmoidf_(acc);
    } else if (i < B_ + 2 * B_ * H_) {
        int k = i - B_;
        out[i] = (k < B_ * H_) ? h0w[k] : h1w[k - B_ * H_];
    }
}

// ---- Host launcher ----------------------------------------------------------
extern "C" void kernel_launch(void* const* d_in, const int* in_sizes, int n_in,
                              void* d_out, int out_size, void* d_ws, size_t ws_size,
                              hipStream_t stream) {
    (void)in_sizes; (void)n_in; (void)out_size; (void)ws_size;
    const int*   words  = (const int*)d_in[0];
    const float* hidden = (const float*)d_in[1];
    const float* emb    = (const float*)d_in[2];
    const float* w_ih0  = (const float*)d_in[3];
    const float* w_hh0  = (const float*)d_in[4];
    const float* b_ih0  = (const float*)d_in[5];
    const float* b_hh0  = (const float*)d_in[6];
    const float* w_ih1  = (const float*)d_in[7];
    const float* w_hh1  = (const float*)d_in[8];
    const float* b_ih1  = (const float*)d_in[9];
    const float* b_hh1  = (const float*)d_in[10];
    const float* fc_w   = (const float*)d_in[11];
    const float* fc_b   = (const float*)d_in[12];
    float* out = (float*)d_out;

    char* wsp = (char*)d_ws;
    size_t o = 0;
    auto alloc = [&](size_t bytes) { void* p = wsp + o; o += (bytes + 255) & ~(size_t)255; return p; };
    bf16*  WIH0F = (bf16*)alloc((size_t)10 * NT_ * 512 * 2);
    bf16*  WHH0F = (bf16*)alloc((size_t)16 * NT_ * 512 * 2);
    bf16*  WIH1F = (bf16*)alloc((size_t)16 * NT_ * 512 * 2);
    bf16*  WHH1F = (bf16*)alloc((size_t)16 * NT_ * 512 * 2);
    bf16*  X0    = (bf16*)alloc((size_t)MP_ * K0P_ * 2);
    bf16*  X1    = (bf16*)alloc((size_t)MP_ * H_ * 2);
    float* XG0   = (float*)alloc((size_t)MP_ * G_ * 4);
    float* XG1   = (float*)alloc((size_t)MP_ * G_ * 4);
    float* H0W   = (float*)alloc((size_t)2 * 32 * H_ * 4);   // ping-pong
    float* H1W   = (float*)alloc((size_t)2 * 32 * H_ * 4);
    bf16*  HB0   = (bf16*)alloc((size_t)2 * 32 * H_ * 2);
    bf16*  HB1   = (bf16*)alloc((size_t)2 * 32 * H_ * 2);
    int*   CNT   = (int*)alloc(256);

    const int SMEM_SCAN = 16 * 12 * 512 * (int)sizeof(bf16)
                        + 32 * 192 * (int)sizeof(float)
                        + 192 * (int)sizeof(float);          // 221952 B
    const int SMEM_GEMM = 16 * 4 * 512 * (int)sizeof(bf16);  // 65536 B (max ktiles)
    hipFuncSetAttribute((const void*)gru_scan_kernel,
                        hipFuncAttributeMaxDynamicSharedMemorySize, SMEM_SCAN);
    hipFuncSetAttribute((const void*)gemm_bias_kernel,
                        hipFuncAttributeMaxDynamicSharedMemorySize, SMEM_GEMM);

    const int mtiles = MP_ / 16;                             // 938
    const int mpgrps = ((mtiles / 2) + 7) / 8;               // 59
    const int gemmBlocks = mpgrps * (NT_ / 4);               // 1416

    int initN = 2 * 2 * 32 * H_ + (MP_ - BT_) * H_ + 64;
    init_state_kernel<<<(initN + 255) / 256, 256, 0, stream>>>(hidden, H0W, H1W, HB0, HB1, X1, CNT);
    convert_weight_kernel<<<(10 * NT_ * 512 + 255) / 256, 256, 0, stream>>>(w_ih0, WIH0F, D_, 10);
    convert_weight_kernel<<<(16 * NT_ * 512 + 255) / 256, 256, 0, stream>>>(w_hh0, WHH0F, H_, 16);
    convert_weight_kernel<<<(16 * NT_ * 512 + 255) / 256, 256, 0, stream>>>(w_ih1, WIH1F, H_, 16);
    convert_weight_kernel<<<(16 * NT_ * 512 + 255) / 256, 256, 0, stream>>>(w_hh1, WHH1F, H_, 16);
    embed_gather_kernel<<<(int)(((long)MP_ * K0P_ + 255) / 256), 256, 0, stream>>>(words, emb, X0);

    gemm_bias_kernel<<<gemmBlocks, 256, 10 * 4 * 512 * 2, stream>>>(X0, K0P_, 10, WIH0F, b_ih0, XG0, mtiles);
    gru_scan_kernel<<<8, 384, SMEM_SCAN, stream>>>(XG0, WHH0F, b_hh0, H0W, HB0, X1, CNT);
    gemm_bias_kernel<<<gemmBlocks, 256, 16 * 4 * 512 * 2, stream>>>(X1, H_, 16, WIH1F, b_ih1, XG1, mtiles);
    gru_scan_kernel<<<8, 384, SMEM_SCAN, stream>>>(XG1, WHH1F, b_hh1, H1W, HB1, nullptr, CNT + 16);

    finalize_kernel<<<(B_ + 2 * B_ * H_ + 255) / 256, 256, 0, stream>>>(H0W, H1W, fc_w, fc_b, out);
}
